// O1Consequent_34600256536768
// MI455X (gfx1250) — compile-verified, matrix operand order checked by reading, not run on hardware
//
#include <hip/hip_runtime.h>

// ---------------------------------------------------------------------------
// out[b,o] = sum_{r,i} x[b,i] * frs[b,r] * W[r,o,i]  +  sum_r frs[b,r]*bias[r,o]
//
// One GEMM, M=8192 N=512 K=8192 (+1 bias K-tile):
//   A'[b, r*512+i] = frs[b,r] * x[b,i]   (scaled+converted to bf16 in LDS)
//   B'[r*512+i, o] = W[r, o, i]          (pre-converted to bf16 in d_ws once,
//                                         then async-copied global->LDS)
// f32 accumulate via v_wmma_f32_16x16x32_bf16.
// ---------------------------------------------------------------------------

typedef __bf16 v16bf __attribute__((ext_vector_type(16)));
typedef float  v8f   __attribute__((ext_vector_type(8)));

#define IN_DIM   512
#define OUT_DIM  512
#define N_RULES  16
#define BATCH    8192

#define BM 128           // block tile M (batch)
#define BN 128           // block tile N (out)
#define KT 32            // K tile (bf16 elems) == one WMMA K
#define LDSS 40          // LDS row stride (KT + 8 pad) -> conflict-free b128
#define NTILES 257       // 256 K tiles + 1 leading bias tile

__device__ __forceinline__ unsigned short f2bf(float f) {
  union { __bf16 h; unsigned short u; } c;
  c.h = (__bf16)f;                 // native v_cvt on gfx1250
  return c.u;
}

// Async global->LDS copy, 16B per lane. Tracked by ASYNCcnt.
__device__ __forceinline__ void async_b128(unsigned lds_off, const void* gptr) {
  asm volatile("global_load_async_to_lds_b128 %0, %1, off"
               :: "v"(lds_off), "v"(gptr) : "memory");
}
__device__ __forceinline__ void wait_async0() {
  asm volatile("s_wait_asynccnt 0x0" ::: "memory");
}

// One-shot W f32 -> bf16 conversion into workspace.
__global__ __launch_bounds__(256)
void convert_w(const float* __restrict__ W, unsigned short* __restrict__ Wbf) {
  const size_t i = ((size_t)blockIdx.x * 256 + threadIdx.x) * 8;
  float4 a = *(const float4*)(W + i);
  float4 b = *(const float4*)(W + i + 4);
  union { unsigned short u[8]; uint4 q; } o;
  o.u[0] = f2bf(a.x); o.u[1] = f2bf(a.y); o.u[2] = f2bf(a.z); o.u[3] = f2bf(a.w);
  o.u[4] = f2bf(b.x); o.u[5] = f2bf(b.y); o.u[6] = f2bf(b.z); o.u[7] = f2bf(b.w);
  *(uint4*)(Wbf + i) = o.q;
}

// Stage one 128x32 A tile (+B tile) into LDS. tile==0 is the bias pseudo-tile.
template <bool WBF>
__device__ __forceinline__ void stage_tile(
    unsigned short* __restrict__ a, unsigned short* __restrict__ bq,
    const float* __restrict__ x, const float* __restrict__ W,
    const unsigned short* __restrict__ Wbf,
    const float* __restrict__ frs, const float* __restrict__ bias,
    const float* __restrict__ sFrs,
    int blockM, int blockN, int tid, int tile) {
  const int row  = tid >> 1;   // 0..127
  const int half = tid & 1;    // k 0..15 / 16..31
  unsigned short va[16];

  if (tile == 0) {
    unsigned short vb[16];
    if (half == 0) {
      const float* fr = frs + (size_t)(blockM + row) * N_RULES;
#pragma unroll
      for (int j = 0; j < 16; ++j) va[j] = f2bf(fr[j]);
      const float* bb = bias + (blockN + row);
#pragma unroll
      for (int j = 0; j < 16; ++j) vb[j] = f2bf(bb[(size_t)j * OUT_DIM]);
    } else {
#pragma unroll
      for (int j = 0; j < 16; ++j) { va[j] = 0; vb[j] = 0; }
    }
    uint4* pa = (uint4*)&a [row * LDSS + half * 16];
    uint4* pb = (uint4*)&bq[row * LDSS + half * 16];
    pa[0] = ((const uint4*)va)[0]; pa[1] = ((const uint4*)va)[1];
    pb[0] = ((const uint4*)vb)[0]; pb[1] = ((const uint4*)vb)[1];
    return;
  }

  const int koff = (tile - 1) * KT;          // multiple of 32
  const int r    = koff >> 9;                // rule index, uniform in tile
  const int io0  = koff & (IN_DIM - 1);
  const int io   = io0 + half * 16;
  const float scale = sFrs[row * N_RULES + r];

  // ---- A: load f32 x, scale by frs[b,r], convert, ds_store ----
  const float4* xs = (const float4*)(x + (size_t)(blockM + row) * IN_DIM + io);
#pragma unroll
  for (int q = 0; q < 4; ++q) {
    float4 v = xs[q];
    va[q*4+0] = f2bf(v.x * scale); va[q*4+1] = f2bf(v.y * scale);
    va[q*4+2] = f2bf(v.z * scale); va[q*4+3] = f2bf(v.w * scale);
  }
  uint4* pa = (uint4*)&a[row * LDSS + half * 16];
  pa[0] = ((const uint4*)va)[0]; pa[1] = ((const uint4*)va)[1];

  if (WBF) {
    // ---- B: pure bf16 copy, async global->LDS (ASYNCcnt), 2 x b128/thread ----
#pragma unroll
    for (int q = 0; q < 2; ++q) {
      const int c   = tid + q * 256;          // 0..511 chunk id (8 bf16 each)
      const int brow = c >> 2;                // 0..127 (out col)
      const int sub  = c & 3;                 // chunk within row
      const unsigned lds_off =
          (unsigned)(size_t)(void*)&bq[brow * LDSS + sub * 8];
      const unsigned short* g = Wbf + (size_t)r * IN_DIM * OUT_DIM +
                                (size_t)(blockN + brow) * IN_DIM + io0 + sub * 8;
      async_b128(lds_off, g);
    }
    if (tile + 1 < NTILES) {
      const int kp = tile * KT;
      const int ip = (kp & (IN_DIM - 1)) + half * 16;
      __builtin_prefetch(x + (size_t)(blockM + row) * IN_DIM + ip, 0, 1);
    }
  } else {
    // ---- B: f32 load + convert fallback ----
    unsigned short vb[16];
    const float4* ws = (const float4*)(W + (size_t)r * IN_DIM * OUT_DIM +
                                       (size_t)(blockN + row) * IN_DIM + io);
#pragma unroll
    for (int q = 0; q < 4; ++q) {
      float4 v = ws[q];
      vb[q*4+0] = f2bf(v.x); vb[q*4+1] = f2bf(v.y);
      vb[q*4+2] = f2bf(v.z); vb[q*4+3] = f2bf(v.w);
    }
    uint4* pb = (uint4*)&bq[row * LDSS + half * 16];
    pb[0] = ((const uint4*)vb)[0]; pb[1] = ((const uint4*)vb)[1];
    if (tile + 1 < NTILES) {
      const int kp = tile * KT;
      const int rp = kp >> 9;
      const int ip = (kp & (IN_DIM - 1)) + half * 16;
      __builtin_prefetch(x + (size_t)(blockM + row) * IN_DIM + ip, 0, 1);
      __builtin_prefetch(W + (size_t)rp * IN_DIM * OUT_DIM +
                         (size_t)(blockN + row) * IN_DIM + ip, 0, 1);
    }
  }
}

// ISA 16-bit A/B fragment layout: lanes 0-15 take K 0..7 & 16..23,
// lanes 16-31 take K 8..15 & 24..31 of their row/col.
__device__ __forceinline__ v16bf load_frag(const unsigned short* __restrict__ base,
                                           int laneMod, int laneHalf) {
  const unsigned short* q = base + laneMod * LDSS + laneHalf * 8;
  union { v16bf v; uint4 q4[2]; } u;
  u.q4[0] = *(const uint4*)(q);
  u.q4[1] = *(const uint4*)(q + 16);
  return u.v;
}

template <bool WBF>
__global__ __launch_bounds__(256)
void fused_rule_gemm(const float* __restrict__ x,
                     const float* __restrict__ frs,
                     const float* __restrict__ W,
                     const unsigned short* __restrict__ Wbf,
                     const float* __restrict__ bias,
                     float* __restrict__ out) {
  __shared__ unsigned short sA[2][BM * LDSS];
  __shared__ unsigned short sB[2][BN * LDSS];
  __shared__ float sFrs[BM * N_RULES];

  const int tid    = threadIdx.x;
  const int blockN = blockIdx.x * BN;
  const int blockM = blockIdx.y * BM;

  const int wave     = tid >> 5;
  const int lane     = tid & 31;
  const int waveM    = wave >> 2;      // 0..1 -> 64-row slice
  const int waveN    = wave & 3;       // 0..3 -> 32-col slice
  const int laneMod  = lane & 15;
  const int laneHalf = (lane >> 4) & 1;

  {  // preload frs block (128 x 16 f32)
    const float4* src = (const float4*)(frs + (size_t)blockM * N_RULES);
    float4* dst = (float4*)sFrs;
    dst[tid]       = src[tid];
    dst[tid + 256] = src[tid + 256];
  }

  v8f acc[4][2] = {};

  stage_tile<WBF>(sA[0], sB[0], x, W, Wbf, frs, bias, sFrs, blockM, blockN, tid, 0);
  if (WBF) wait_async0();
  __syncthreads();

  for (int t = 0; t < NTILES; ++t) {
    const int cur = t & 1;
    if (t + 1 < NTILES)
      stage_tile<WBF>(sA[cur ^ 1], sB[cur ^ 1], x, W, Wbf, frs, bias, sFrs,
                      blockM, blockN, tid, t + 1);

    const unsigned short* aT = sA[cur];
    const unsigned short* bT = sB[cur];

    v16bf afr[4];
#pragma unroll
    for (int mf = 0; mf < 4; ++mf)
      afr[mf] = load_frag(aT + (waveM * 64 + mf * 16) * LDSS, laneMod, laneHalf);
    v16bf bfr[2];
#pragma unroll
    for (int nf = 0; nf < 2; ++nf)
      bfr[nf] = load_frag(bT + (waveN * 32 + nf * 16) * LDSS, laneMod, laneHalf);

#pragma unroll
    for (int mf = 0; mf < 4; ++mf)
#pragma unroll
      for (int nf = 0; nf < 2; ++nf)
        acc[mf][nf] = __builtin_amdgcn_wmma_f32_16x16x32_bf16(
            false, afr[mf], false, bfr[nf], (short)0, acc[mf][nf], false, false);

    if (WBF) wait_async0();   // own wave's async B copies done before handoff
    __syncthreads();
  }

  // C/D layout: VGPR j -> M = j (lanes 0-15) / j+8 (lanes 16-31); N = lane&15.
#pragma unroll
  for (int mf = 0; mf < 4; ++mf) {
#pragma unroll
    for (int nf = 0; nf < 2; ++nf) {
      const int orow0 = blockM + waveM * 64 + mf * 16 + laneHalf * 8;
      const int ocol  = blockN + waveN * 32 + nf * 16 + laneMod;
#pragma unroll
      for (int j = 0; j < 8; ++j)
        out[(size_t)(orow0 + j) * OUT_DIM + ocol] = acc[mf][nf][j];
    }
  }
}

extern "C" void kernel_launch(void* const* d_in, const int* in_sizes, int n_in,
                              void* d_out, int out_size, void* d_ws, size_t ws_size,
                              hipStream_t stream) {
  const float* x    = (const float*)d_in[0];  // (8192, 512)
  const float* frs  = (const float*)d_in[1];  // (8192, 16)
  const float* W    = (const float*)d_in[2];  // (16, 512, 512)
  const float* bias = (const float*)d_in[3];  // (16, 512)
  float* out        = (float*)d_out;          // (8192, 512)

  const size_t WELEMS = (size_t)N_RULES * IN_DIM * OUT_DIM;   // 4,194,304
  dim3 grid(OUT_DIM / BN, BATCH / BM);                        // (4, 64)

  if (ws_size >= WELEMS * sizeof(unsigned short)) {
    unsigned short* Wbf = (unsigned short*)d_ws;
    convert_w<<<(unsigned)(WELEMS / (256 * 8)), 256, 0, stream>>>(W, Wbf);
    fused_rule_gemm<true><<<grid, 256, 0, stream>>>(x, frs, W, Wbf, bias, out);
  } else {
    fused_rule_gemm<false><<<grid, 256, 0, stream>>>(x, frs, W, nullptr, bias, out);
  }
}
